// DFNBokehModel_87411174409120
// MI455X (gfx1250) — compile-verified, matrix operand order checked by reading, not run on hardware
//
#include <hip/hip_runtime.h>

// ---------------------------------------------------------------------------
// DFN bokeh pipeline for MI455X (gfx1250, wave32, WMMA + TDM).
// Convs = implicit GEMM on v_wmma_f32_16x16x32_bf16 with tap-major K ordering
//   k = (dy*3+dx)*Cin + ci, activations NHWC bf16.
// A fragments: two contiguous 16B LDS loads; B fragments: two global_load_b128
// from prepacked (L2-resident) weights; each wave: 1 M-tile x 2 N-tiles.
// LDS halo staging uses the Tensor Data Mover (tensor_load_to_lds) with D#
// padding fields producing the bank-conflict-free pixel stride; boundary
// blocks fall back to manual staging (zero pad).
// ---------------------------------------------------------------------------

typedef __attribute__((ext_vector_type(16))) __bf16 v16bf;
typedef __attribute__((ext_vector_type(8)))  float  v8f;
typedef __attribute__((ext_vector_type(4))) unsigned int u32x4;
typedef __attribute__((ext_vector_type(8))) int          i32x8;
typedef __attribute__((ext_vector_type(4))) int          i32x4;

#define HH 512
#define WW 512
#define BB 4
#define HW (HH * WW)

#ifndef __has_builtin
#define __has_builtin(x) 0
#endif
#if __has_builtin(__builtin_amdgcn_tensor_load_to_lds) && \
    __has_builtin(__builtin_amdgcn_s_wait_tensorcnt)
#define USE_TDM 1
#else
#define USE_TDM 0
#endif

// ---------- pack rgb+depth -> bf16 NHWC [B,H,W,4] ----------
__global__ void pack_input_kernel(const float* __restrict__ rgb,
                                  const float* __restrict__ depth,
                                  __bf16* __restrict__ x0) {
  int idx = blockIdx.x * blockDim.x + threadIdx.x;
  const int total = BB * HW * 4;
  if (idx >= total) return;
  int c = idx & 3;
  int t = idx >> 2;
  int p = t % HW;
  int b = t / HW;
  float v = (c < 3) ? rgb[((size_t)(b * 3 + c)) * HW + p]
                    : depth[(size_t)b * HW + p];
  x0[idx] = (__bf16)v;
}

// ---------- pack OIHW f32 weights into WMMA B-fragment layout ----------
// K order: k = tap*Cin + ci  (tap = dy*3+dx).
__global__ void pack_weights_kernel(const float* __restrict__ w,
                                    __bf16* __restrict__ wp,
                                    int Cout, int Cin, int ntAlloc, int ksteps) {
  int idx = blockIdx.x * blockDim.x + threadIdx.x;
  int total = ntAlloc * ksteps * 32;
  if (idx >= total) return;
  int lane = idx & 31;
  int t    = idx >> 5;
  int ks   = t % ksteps;
  int nt   = t / ksteps;
  int n     = lane & 15;
  int khalf = lane >> 4;
  int co    = nt * 16 + n;
  int Kmax  = Cin * 9;
  __bf16* dst = wp + (size_t)idx * 16;
#pragma unroll
  for (int j = 0; j < 16; ++j) {
    int k = ks * 32 + khalf * 8 + ((j < 8) ? j : (j + 8));
    float f = 0.0f;
    if (co < Cout && k < Kmax) {
      int tap = k / Cin;
      int ci  = k - tap * Cin;
      f = w[((size_t)co * Cin + ci) * 9 + tap];   // OIHW, tap = dy*3+dx
    }
    dst[j] = (__bf16)f;
  }
}

// ---------- 3x3 SAME conv (zero pad) + bias (+ReLU) via WMMA ----------
// in/out NHWC bf16. Block: 256 thr = 8 waves = 4 M-tiles (64 px) x 2 N-groups
// (each = 2 N-tiles = 32 ch). grid: (W/64, H, B*cblks), cblks = ntAlloc/4.
template <int CIN>
__global__ __launch_bounds__(256)
void conv3x3_wmma_kernel(const __bf16* __restrict__ in,
                         const __bf16* __restrict__ wp,
                         const float*  __restrict__ bias,
                         __bf16* __restrict__ out,
                         int Cout, int cblks, int do_relu) {
  constexpr int K      = CIN * 9;
  constexpr int KSTEPS = (K + 31) / 32;
  constexpr bool FAST  = (CIN % 32 == 0);
  constexpr int CINP   = FAST ? (CIN + 8) : CIN;   // LDS pixel stride

  __shared__ __bf16 s_in[3 * 66 * CINP];

  const int b    = blockIdx.z / cblks;
  const int cblk = blockIdx.z % cblks;
  const int y    = blockIdx.y;
  const int x0   = blockIdx.x * 64;

  // ---- stage NHWC halo patch (rows y-1..y+1, cols x0-1..x0+64) ----
  const bool interior = (y >= 1) && (y <= HH - 2) && (x0 >= 1) && (x0 + 65 < WW);
#if USE_TDM
  if (interior) {
    // Tensor Data Mover: 2D tile, 3 rows of 66*CIN contiguous bf16 elements,
    // row stride WW*CIN; LDS padding gives pixel stride CINP.
    if (threadIdx.x < 32) {
      const __bf16* src = in + (size_t)((b * HH + (y - 1)) * WW + (x0 - 1)) * CIN;
      uint64_t ga      = (uint64_t)(uintptr_t)src;
      uint32_t lds_off = (uint32_t)(uintptr_t)(&s_in[0]);  // LDS aperture: low 32 bits
      uint32_t tdim0   = (uint32_t)(WW - (x0 - 1)) * CIN;  // valid elems from tile start
      uint32_t tdim1   = 3;
      uint32_t tile0   = 66 * CIN;
      uint32_t stride0 = WW * CIN;

      u32x4 g0;
      g0[0] = 1u;                                   // count=1, user descriptor
      g0[1] = lds_off;                              // [63:32] lds_addr
      g0[2] = (uint32_t)ga;                         // [95:64] global_addr lo
      g0[3] = (uint32_t)((ga >> 32) & 0x1FFFFFFu)   // [120:96] global_addr hi
              | (2u << 30);                         // [127:126] type=2 (image)

      uint32_t d0 = (1u << 16);                     // data_size = 2 bytes
      if (FAST) {
        const uint32_t pint = (CIN == 64) ? 4u : 5u;  // 32 / 64 DWORDs per pixel
        d0 |= (1u << 20) | (pint << 22) | (3u << 25); // pad_enable, +4 DWORDs
      }
      i32x8 g1;
      g1[0] = (int)d0;
      g1[1] = (int)((tdim0 & 0xFFFFu) << 16);                         // tensor_dim0[15:0] @48
      g1[2] = (int)(((tdim0 >> 16) & 0xFFFFu) | ((tdim1 & 0xFFFFu) << 16));
      g1[3] = (int)(((tdim1 >> 16) & 0xFFFFu) | ((tile0 & 0xFFFFu) << 16)); // tile_dim0 @112
      g1[4] = (int)3;                                                 // tile_dim1=3, tile_dim2=0
      g1[5] = (int)stride0;                                           // tensor_dim0_stride lo
      g1[6] = 0;                                                      // stride hi / dim1_stride
      g1[7] = 0;
      i32x4 g2 = {0, 0, 0, 0};
      i32x4 g3 = {0, 0, 0, 0};
      i32x8 g4 = {0, 0, 0, 0, 0, 0, 0, 0};          // trailing operand (clang-23 form)
      __builtin_amdgcn_tensor_load_to_lds(g0, g1, g2, g3, g4, 0);
      __builtin_amdgcn_s_wait_tensorcnt(0);
    }
  } else
#endif
  {
    constexpr int C2 = CIN / 2;                 // u32 per pixel
    const uint32_t* src32 = (const uint32_t*)in;
    const int total32 = 3 * 66 * C2;
    for (int i = threadIdx.x; i < total32; i += 256) {
      int pi = i / C2;
      int cp = i - pi * C2;
      int ry = pi / 66;
      int sx = pi - ry * 66;
      int gy = y + ry - 1;
      int gx = x0 + sx - 1;
      uint32_t v = 0;
      if (gy >= 0 && gy < HH && gx >= 0 && gx < WW)
        v = src32[(size_t)((b * HH + gy) * WW + gx) * C2 + cp];
      *(uint32_t*)&s_in[(ry * 66 + sx) * CINP + 2 * cp] = v;
    }
  }
  __syncthreads();

  const int lane  = threadIdx.x & 31;
  const int wv    = threadIdx.x >> 5;
  const int mtile = wv & 3;            // 16-pixel sub-strip
  const int ng    = wv >> 2;           // N-group (2 ntiles each)
  const int nt0   = cblk * 4 + ng * 2;
  const int row   = lane & 15;
  const int khalf = lane >> 4;
  const int px    = mtile * 16 + row;

  v8f acc0 = {}, acc1 = {};
  const __bf16* wp0 = wp + ((size_t)nt0 * KSTEPS * 32 + lane) * 16;
  const __bf16* wp1 = wp0 + (size_t)KSTEPS * 32 * 16;

  for (int ks = 0; ks < KSTEPS; ++ks) {
    // B fragments: contiguous 32B per lane
    v16bf bf0, bf1;
    ((uint4*)&bf0)[0] = *(const uint4*)(wp0 + ks * 512);
    ((uint4*)&bf0)[1] = *(const uint4*)(wp0 + ks * 512 + 8);
    ((uint4*)&bf1)[0] = *(const uint4*)(wp1 + ks * 512);
    ((uint4*)&bf1)[1] = *(const uint4*)(wp1 + ks * 512 + 8);

    // A fragment
    v16bf af;
    if constexpr (FAST) {
      constexpr int CHUNKS_PER_TAP = CIN / 32;
      int tap    = ks / CHUNKS_PER_TAP;
      int cibase = (ks - tap * CHUNKS_PER_TAP) * 32;
      int dy = tap / 3, dx = tap - dy * 3;
      const __bf16* base = &s_in[(dy * 66 + px + dx) * CINP + cibase + khalf * 8];
      ((uint4*)&af)[0] = *(const uint4*)(base);
      ((uint4*)&af)[1] = *(const uint4*)(base + 16);
    } else {
#pragma unroll
      for (int j = 0; j < 16; ++j) {
        int k   = ks * 32 + khalf * 8 + ((j < 8) ? j : (j + 8));
        int tap = k / CIN;
        int ci  = k - tap * CIN;
        int tc  = (tap > 8) ? 8 : tap;
        int dy = tc / 3, dx = tc - dy * 3;
        __bf16 v = s_in[(dy * 66 + px + dx) * CINP + ci];
        af[j] = (k < K) ? v : (__bf16)0.0f;
      }
    }

    acc0 = __builtin_amdgcn_wmma_f32_16x16x32_bf16(false, af, false, bf0,
                                                   (short)0, acc0, false, false);
    acc1 = __builtin_amdgcn_wmma_f32_16x16x32_bf16(false, af, false, bf1,
                                                   (short)0, acc1, false, false);
  }

  // ---- epilogue: D layout — VGPR r: lanes0-15 (M=r,N=lane), lanes16-31 M=r+8
  const int xb = x0 + mtile * 16;
#pragma unroll
  for (int g = 0; g < 2; ++g) {
    const v8f& acc = g ? acc1 : acc0;
    int co = (nt0 + g) * 16 + row;
    if (co < Cout) {
      float bv = bias[co];
#pragma unroll
      for (int r = 0; r < 8; ++r) {
        float v = acc[r] + bv;
        if (do_relu) v = fmaxf(v, 0.0f);
        int xo = xb + r + khalf * 8;
        out[(size_t)((b * HH + y) * WW + xo) * Cout + co] = (__bf16)v;
      }
    }
  }
}

// ---------- per-pixel softmax over 11 channels (NHWC) ----------
__global__ void softmax11_kernel(const __bf16* __restrict__ logits,
                                 float* __restrict__ probs) {
  int idx = blockIdx.x * blockDim.x + threadIdx.x;
  const int total = BB * HW;
  if (idx >= total) return;
  size_t base = (size_t)idx * 11;
  float v[11];
  float m = -1e30f;
#pragma unroll
  for (int k = 0; k < 11; ++k) {
    v[k] = (float)logits[base + k];
    m = fmaxf(m, v[k]);
  }
  float s = 0.0f;
#pragma unroll
  for (int k = 0; k < 11; ++k) { v[k] = __expf(v[k] - m); s += v[k]; }
  float inv = 1.0f / s;
#pragma unroll
  for (int k = 0; k < 11; ++k) probs[base + k] = v[k] * inv;
}

// ---------- horizontal separable filter (edge pad), kh NHWC ----------
__global__ void filter_h_kernel(const float* __restrict__ rgb,
                                const float* __restrict__ kh,
                                float* __restrict__ outh) {
  int idx = blockIdx.x * blockDim.x + threadIdx.x;
  const int total = BB * 3 * HW;
  if (idx >= total) return;
  int p = idx % HW;
  int c = (idx / HW) % 3;
  int b = idx / (3 * HW);
  int y = p / WW;
  int x = p % WW;
  const float* rrow = rgb + ((size_t)(b * 3 + c)) * HW + (size_t)y * WW;
  const float* kp   = kh + ((size_t)b * HW + p) * 11;
  __builtin_prefetch(rrow + x, 0, 0);   // global_prefetch_b8
  float acc = 0.0f;
#pragma unroll
  for (int k = 0; k < 11; ++k) {
    int xs = x + k - 5;
    xs = (xs < 0) ? 0 : ((xs > WW - 1) ? WW - 1 : xs);
    acc += rrow[xs] * kp[k];
  }
  outh[idx] = acc;
}

// ---------- vertical separable filter + depth-mask blend, kv NHWC ----------
__global__ void filter_v_blend_kernel(const float* __restrict__ rgb,
                                      const float* __restrict__ depth,
                                      const float* __restrict__ outh,
                                      const float* __restrict__ kv,
                                      float* __restrict__ out) {
  int idx = blockIdx.x * blockDim.x + threadIdx.x;
  const int total = BB * 3 * HW;
  if (idx >= total) return;
  int p = idx % HW;
  int c = (idx / HW) % 3;
  int b = idx / (3 * HW);
  int y = p / WW;
  int x = p % WW;
  const float* kp = kv + ((size_t)b * HW + p) * 11;
  float acc = 0.0f;
#pragma unroll
  for (int k = 0; k < 11; ++k) {
    int ys = y + k - 5;
    ys = (ys < 0) ? 0 : ((ys > HH - 1) ? HH - 1 : ys);
    acc += outh[((size_t)(b * 3 + c) * HH + ys) * WW + x] * kp[k];
  }
  float d    = depth[(size_t)b * HW + p];
  float mask = 1.0f / (1.0f + __expf(-(d - 0.2f) * 15.0f));
  float r    = rgb[idx];
  out[idx]   = mask * r + (1.0f - mask) * acc;
}

// ---------------------------------------------------------------------------
extern "C" void kernel_launch(void* const* d_in, const int* in_sizes, int n_in,
                              void* d_out, int out_size, void* d_ws, size_t ws_size,
                              hipStream_t stream) {
  (void)in_sizes; (void)n_in; (void)out_size; (void)ws_size;

  const float* rgb   = (const float*)d_in[0];
  const float* depth = (const float*)d_in[1];
  const float* w[7]  = {(const float*)d_in[2],  (const float*)d_in[4],
                        (const float*)d_in[6],  (const float*)d_in[8],
                        (const float*)d_in[10], (const float*)d_in[12],
                        (const float*)d_in[14]};
  const float* bs[7] = {(const float*)d_in[3],  (const float*)d_in[5],
                        (const float*)d_in[7],  (const float*)d_in[9],
                        (const float*)d_in[11], (const float*)d_in[13],
                        (const float*)d_in[15]};
  const int cin[7]  = {4, 64, 64, 128, 128, 64, 64};
  const int cout[7] = {64, 64, 128, 128, 64, 11, 11};

  // Carve workspace
  char* p = (char*)d_ws;
  auto alloc = [&](size_t bytes) -> char* {
    char* r = p;
    p += (bytes + 255) & ~(size_t)255;
    return r;
  };
  __bf16* x0   = (__bf16*)alloc((size_t)BB * HW * 4   * 2);  // NHWC
  __bf16* actA = (__bf16*)alloc((size_t)BB * HW * 128 * 2);
  __bf16* actB = (__bf16*)alloc((size_t)BB * HW * 128 * 2);
  __bf16* khl  = (__bf16*)alloc((size_t)BB * HW * 11  * 2);
  __bf16* kvl  = (__bf16*)alloc((size_t)BB * HW * 11  * 2);
  float*  khp  = (float*)alloc((size_t)BB * HW * 11 * 4);
  float*  kvp  = (float*)alloc((size_t)BB * HW * 11 * 4);
  float*  outh = (float*)alloc((size_t)BB * 3 * HW * 4);     // NCHW

  __bf16* wp[7];
  int ntA[7], kst[7];
  for (int l = 0; l < 7; ++l) {
    int nt = (cout[l] + 15) / 16;
    ntA[l] = (nt + 3) & ~3;                  // multiple of 4 (2 N-groups/block)
    kst[l] = (cin[l] * 9 + 31) / 32;
    wp[l]  = (__bf16*)alloc((size_t)ntA[l] * kst[l] * 32 * 16 * 2);
  }

  // 1) pack input + weights
  {
    int total = BB * HW * 4;
    pack_input_kernel<<<(total + 255) / 256, 256, 0, stream>>>(rgb, depth, x0);
  }
  for (int l = 0; l < 7; ++l) {
    int total = ntA[l] * kst[l] * 32;
    pack_weights_kernel<<<(total + 63) / 64, 64, 0, stream>>>(
        w[l], wp[l], cout[l], cin[l], ntA[l], kst[l]);
  }

  // 2) conv stack
  auto conv = [&](const __bf16* in, int l, __bf16* o, int relu) {
    int cblks = ntA[l] / 4;
    dim3 g(WW / 64, HH, BB * cblks);
    if (cin[l] == 4)
      conv3x3_wmma_kernel<4><<<g, 256, 0, stream>>>(in, wp[l], bs[l], o,
                                                    cout[l], cblks, relu);
    else if (cin[l] == 64)
      conv3x3_wmma_kernel<64><<<g, 256, 0, stream>>>(in, wp[l], bs[l], o,
                                                     cout[l], cblks, relu);
    else
      conv3x3_wmma_kernel<128><<<g, 256, 0, stream>>>(in, wp[l], bs[l], o,
                                                      cout[l], cblks, relu);
  };
  conv(x0,   0, actA, 1);   // 4   -> 64
  conv(actA, 1, actB, 1);   // 64  -> 64
  conv(actB, 2, actA, 1);   // 64  -> 128
  conv(actA, 3, actB, 1);   // 128 -> 128
  conv(actB, 4, actA, 1);   // 128 -> 64
  conv(actA, 5, khl,  0);   // 64  -> 11 (logits)
  conv(actA, 6, kvl,  0);   // 64  -> 11 (logits)

  // 3) softmax + separable filters + blend
  {
    int total = BB * HW;
    softmax11_kernel<<<(total + 255) / 256, 256, 0, stream>>>(khl, khp);
    softmax11_kernel<<<(total + 255) / 256, 256, 0, stream>>>(kvl, kvp);
  }
  {
    int total = BB * 3 * HW;
    filter_h_kernel<<<(total + 255) / 256, 256, 0, stream>>>(rgb, khp, outh);
    filter_v_blend_kernel<<<(total + 255) / 256, 256, 0, stream>>>(
        rgb, depth, outh, kvp, (float*)d_out);
  }
}
